// Model_56513179681207
// MI455X (gfx1250) — matrix operand order repaired and checked by reading, not compiled, not run on hardware
//
#include <hip/hip_runtime.h>
#include <hip/hip_bf16.h>
#include <stdint.h>
#include <stddef.h>

// ---------------------------------------------------------------------------
// Types for CDNA5 WMMA / TDM
// ---------------------------------------------------------------------------
typedef __attribute__((ext_vector_type(16))) _Float16 v16h;
typedef __attribute__((ext_vector_type(8)))  float    v8f;
typedef __attribute__((ext_vector_type(2)))  float    v2f;
typedef __attribute__((ext_vector_type(4)))  unsigned int u32x4;
typedef __attribute__((ext_vector_type(4)))  int      i32x4;
typedef __attribute__((ext_vector_type(8)))  int      i32x8;

union HalfVec { v16h v; uint4 q[2]; };

#if defined(__has_builtin)
#if __has_builtin(__builtin_amdgcn_wmma_f32_16x16x4_f32)
#define HAS_WMMA_F32 1
#endif
#if __has_builtin(__builtin_amdgcn_tensor_load_to_lds) && __has_builtin(__builtin_amdgcn_s_wait_tensorcnt)
#define HAS_TDM 1
#endif
#endif

#ifndef HAS_WMMA_F32
#warning "CDNA5 probe: __builtin_amdgcn_wmma_f32_16x16x4_f32 NOT available - AtA uses VALU fallback"
#endif
#ifndef HAS_TDM
#warning "CDNA5 probe: tensor_load_to_lds builtin NOT available"
#endif

// Problem constants
#define PADW  132   // 128 + 2*2 halo
#define ACT_HALFS (PADW*PADW*64)

// ---------------------------------------------------------------------------
// d1 stencil coefficient (n = 32, h = 0.01)
// ---------------------------------------------------------------------------
__device__ __forceinline__ float d1c(int i, int j) {
  const float inv_h = 100.0f;
  if (i == 0)  return (j == 0)  ? -inv_h : ((j == 1)  ? inv_h : 0.0f);
  if (i == 31) return (j == 30) ? -inv_h : ((j == 31) ? inv_h : 0.0f);
  if (j == i - 1) return -0.5f * inv_h;
  if (j == i + 1) return  0.5f * inv_h;
  return 0.0f;
}

// ---------------------------------------------------------------------------
// Clear padded activation buffers (borders must be zero for implicit padding)
// ---------------------------------------------------------------------------
__global__ void clear_half_kernel(_Float16* p, int n) {
  int i = blockIdx.x * blockDim.x + threadIdx.x;
  if (i < n) p[i] = (_Float16)0.0f;
}

// ---------------------------------------------------------------------------
// Conv layer 0: Cin=1 -> Cout=64, 5x5, pad 2, ReLU.  VALU (52 MFLOP).
// ---------------------------------------------------------------------------
__global__ __launch_bounds__(256) void conv0_kernel(
    const float* __restrict__ u, const float* __restrict__ W,
    const float* __restrict__ bias, _Float16* __restrict__ out) {
  int idx = blockIdx.x * blockDim.x + threadIdx.x;   // 128*128*64
  int co = idx & 63;
  int p  = idx >> 6;
  int x  = p & 127;
  int y  = p >> 7;
  float acc = bias[co];
  const float* w = W + co * 25;
  #pragma unroll
  for (int dy = 0; dy < 5; ++dy) {
    int yy = y + dy - 2;
    if (yy < 0 || yy > 127) continue;
    #pragma unroll
    for (int dx = 0; dx < 5; ++dx) {
      int xx = x + dx - 2;
      if (xx < 0 || xx > 127) continue;
      acc = fmaf(w[dy * 5 + dx], u[yy * 128 + xx], acc);
    }
  }
  acc = fmaxf(acc, 0.0f);
  out[((y + 2) * PADW + (x + 2)) * 64 + co] = (_Float16)acc;
}

// ---------------------------------------------------------------------------
// Re-swizzle a (64,64,5,5) f32 weight into WMMA-B lane order, f16.
// ---------------------------------------------------------------------------
__global__ __launch_bounds__(256) void prep_w_kernel(
    const float* __restrict__ W, _Float16* __restrict__ Bprep) {
  int idx = blockIdx.x * blockDim.x + threadIdx.x;   // 50*4*32*16 = 102400
  if (idx >= 102400) return;
  int t    = idx & 15;
  int lane = (idx >> 4) & 31;
  int nt   = (idx >> 9) & 3;
  int kb   = idx >> 11;            // 0..49
  int tap  = kb >> 1;
  int half = kb & 1;
  int dy = tap / 5, dx = tap - dy * 5;
  int kk = ((t & 8) << 1) + ((lane < 16) ? 0 : 8) + (t & 7);
  int ci = half * 32 + kk;
  int co = nt * 16 + (lane & 15);
  Bprep[idx] = (_Float16)W[((co * 64 + ci) * 5 + dy) * 5 + dx];
}

// ---------------------------------------------------------------------------
// Conv layers 1..5: 64->64, 5x5, pad 2, ReLU.  Implicit GEMM on WMMA f16.
// ---------------------------------------------------------------------------
__global__ __launch_bounds__(128) void conv_wmma_kernel(
    const _Float16* __restrict__ in, const _Float16* __restrict__ Bprep,
    const float* __restrict__ bias, _Float16* __restrict__ out) {
  int wave = threadIdx.x >> 5;
  int lane = threadIdx.x & 31;
  int ln   = lane & 15;
  bool hi  = lane >= 16;

  int seg = blockIdx.x;            // 0..255
  int y   = seg >> 1;
  int x0  = (seg & 1) * 64 + wave * 16;

  v8f acc0 = {}, acc1 = {}, acc2 = {}, acc3 = {};
  int aoff = hi ? 8 : 0;

  for (int tap = 0; tap < 25; ++tap) {
    int dy = tap / 5, dx = tap - dy * 5;
    const _Float16* src = in + ((size_t)(y + dy) * PADW + (x0 + dx + ln)) * 64;
    #pragma unroll
    for (int half = 0; half < 2; ++half) {
      int c0 = half * 32;
      HalfVec a;
      a.q[0] = *(const uint4*)(src + c0 + aoff);
      a.q[1] = *(const uint4*)(src + c0 + 16 + aoff);
      int kb = tap * 2 + half;
      const _Float16* bbase = Bprep + ((size_t)(kb * 4) * 32 + lane) * 16;
      HalfVec w0, w1, w2, w3;
      w0.q[0] = *(const uint4*)(bbase);            w0.q[1] = *(const uint4*)(bbase + 8);
      w1.q[0] = *(const uint4*)(bbase + 512);      w1.q[1] = *(const uint4*)(bbase + 520);
      w2.q[0] = *(const uint4*)(bbase + 1024);     w2.q[1] = *(const uint4*)(bbase + 1032);
      w3.q[0] = *(const uint4*)(bbase + 1536);     w3.q[1] = *(const uint4*)(bbase + 1544);
      acc0 = __builtin_amdgcn_wmma_f32_16x16x32_f16(false, a.v, false, w0.v, (short)0, acc0, false, false);
      acc1 = __builtin_amdgcn_wmma_f32_16x16x32_f16(false, a.v, false, w1.v, (short)0, acc1, false, false);
      acc2 = __builtin_amdgcn_wmma_f32_16x16x32_f16(false, a.v, false, w2.v, (short)0, acc2, false, false);
      acc3 = __builtin_amdgcn_wmma_f32_16x16x32_f16(false, a.v, false, w3.v, (short)0, acc3, false, false);
    }
  }

  int mbase = hi ? 8 : 0;
  v8f accs[4] = {acc0, acc1, acc2, acc3};
  #pragma unroll
  for (int nt = 0; nt < 4; ++nt) {
    int co = nt * 16 + ln;
    float bv = bias[co];
    #pragma unroll
    for (int r = 0; r < 8; ++r) {
      int m = mbase + r;                                  // pixel x0+m
      float val = fmaxf(accs[nt][r] + bv, 0.0f);
      out[((size_t)(y + 2) * PADW + (x0 + m + 2)) * 64 + co] = (_Float16)val;
    }
  }
}

// ---------------------------------------------------------------------------
// Conv layer 6: 64 -> 1, no ReLU; up = u + conv.
// ---------------------------------------------------------------------------
__global__ __launch_bounds__(256) void conv_last_kernel(
    const _Float16* __restrict__ in, const float* __restrict__ W,
    const float* __restrict__ bias, const float* __restrict__ u,
    float* __restrict__ up_ws, float* __restrict__ up_out) {
  int idx = blockIdx.x * blockDim.x + threadIdx.x;   // 16384
  int x = idx & 127, y = idx >> 7;
  float acc = bias[0];
  for (int ci = 0; ci < 64; ++ci) {
    const float* w = W + ci * 25;
    #pragma unroll
    for (int dy = 0; dy < 5; ++dy) {
      #pragma unroll
      for (int dx = 0; dx < 5; ++dx) {
        acc = fmaf(w[dy * 5 + dx],
                   (float)in[((size_t)(y + dy) * PADW + (x + dx)) * 64 + ci], acc);
      }
    }
  }
  float v = u[idx] + acc;
  up_ws[idx] = v;
  up_out[idx] = v;
}

// ---------------------------------------------------------------------------
// MLP layer (tiny, batch 1)
// ---------------------------------------------------------------------------
__global__ void mlp_layer_kernel(const float* __restrict__ in,
                                 const float* __restrict__ W,
                                 const float* __restrict__ b,
                                 float* __restrict__ out,
                                 int nin, int nout, int do_relu) {
  int o = blockIdx.x * blockDim.x + threadIdx.x;
  if (o >= nout) return;
  float acc = b[o];
  const float* row = W + (size_t)o * nin;
  for (int j = 0; j < nin; ++j) acc = fmaf(row[j], in[j], acc);
  out[o] = (do_relu && acc < 0.0f) ? 0.0f : acc;
}

__global__ void params_store_kernel(const float* __restrict__ pc,
                                    const float* __restrict__ qc,
                                    float* __restrict__ out) {
  int t = threadIdx.x;
  if (t < 3) { out[49153 + t] = pc[t]; out[49156 + t] = qc[t]; }
}

// ---------------------------------------------------------------------------
// p, q, rhs per patch point.
// ---------------------------------------------------------------------------
__global__ __launch_bounds__(256) void build_pq_kernel(
    const float* __restrict__ up, const float* __restrict__ up2,
    const float* __restrict__ pc, const float* __restrict__ qc,
    float* __restrict__ P, float* __restrict__ Q, float* __restrict__ R) {
  int idx = blockIdx.x * blockDim.x + threadIdx.x;   // 16*1024
  int b = idx >> 10, g = idx & 1023;
  int ti = b >> 2, xi = b & 3;
  int pt = g >> 5, px = g & 31;
  int y = ti * 32 + pt, x = xi * 32 + px;
  float uf = up[y * 128 + x];
  float u2 = up2[y * 128 + x];
  P[idx] = pc[0] + pc[1] * uf + pc[2] * uf * uf;
  Q[idx] = qc[0] + qc[1] * u2 + qc[2] * u2 * u2;
  bool bc = (pt == 0) || (pt == 31) || (px == 0) || (px == 31);
  R[idx] = bc ? 10000.0f * uf : 0.0f;
}

// ---------------------------------------------------------------------------
// Dense A build.
// ---------------------------------------------------------------------------
__global__ __launch_bounds__(256) void build_A_kernel(
    const float* __restrict__ P, const float* __restrict__ Q,
    float* __restrict__ A) {
  size_t idx = (size_t)blockIdx.x * blockDim.x + threadIdx.x;  // 16*1024*1024
  int h = (int)(idx & 1023);
  int g = (int)((idx >> 10) & 1023);
  int b = (int)(idx >> 20);
  int gt = g >> 5, gx = g & 31;
  int ht = h >> 5, hx = h & 31;
  float dt  = d1c(gt, ht);
  float dxv = d1c(gx, hx);
  float p = P[b * 1024 + g];
  float q = Q[b * 1024 + g];
  float val = dt * ((gx == hx) ? 1.0f : 0.0f)
            + p * ((gt == ht) ? 1.0f : 0.0f) * dxv
            + q * dt * dxv;
  A[idx] = val;
}

// ---------------------------------------------------------------------------
// M = A^T A + 1e4*diag(bc) + 1e-6*I  via V_WMMA_F32_16X16X4_F32.
// Block = 4 waves; each wave owns one j-tile x FOUR adjacent k-tiles, so the
// transposed A-operand is loaded once per K-chunk and reused by 4 WMMAs.
// Grid = 16 batches * 64 j-tiles * 4 k-groups = 4096 blocks.
// ---------------------------------------------------------------------------
__global__ __launch_bounds__(128) void ata_wmma_kernel(
    const float* __restrict__ A, float* __restrict__ M) {
  int wave = threadIdx.x >> 5;
  int lane = threadIdx.x & 31;
  int ln   = lane & 15;

  int bid = blockIdx.x;
  int b   = bid >> 8;            // / 256
  int rem = bid & 255;
  int jt  = rem >> 2;            // 0..63
  int ktg = rem & 3;
  int kt0 = ktg * 16 + wave * 4; // first of 4 k-tiles for this wave
  int j0  = jt * 16;
  const float* Ab = A + ((size_t)b << 20);

  __builtin_prefetch(Ab + j0, 0, 1);
  __builtin_prefetch(Ab + kt0 * 16, 0, 1);

  float accv[4][8];
#ifdef HAS_WMMA_F32
  v8f acc[4] = {};
  int c0 = (lane < 16) ? 0 : 2;
  for (int i = 0; i < 1024; i += 4) {
    const float* r0 = Ab + (size_t)(i + c0) * 1024;
    const float* r1 = r0 + 1024;
    v2f av;
    av.x = r0[j0 + ln];  av.y = r1[j0 + ln];
    #pragma unroll
    for (int s = 0; s < 4; ++s) {
      int k0 = (kt0 + s) * 16;
      v2f bv;
      bv.x = r0[k0 + ln];  bv.y = r1[k0 + ln];
      acc[s] = __builtin_amdgcn_wmma_f32_16x16x4_f32(false, av, false, bv,
                                                     (short)0, acc[s], false, false);
    }
  }
  #pragma unroll
  for (int s = 0; s < 4; ++s)
    #pragma unroll
    for (int r = 0; r < 8; ++r) accv[s][r] = acc[s][r];
#else
  #pragma unroll
  for (int s = 0; s < 4; ++s)
    #pragma unroll
    for (int r = 0; r < 8; ++r) accv[s][r] = 0.0f;
  int joff = (lane < 16) ? 0 : 8;
  for (int i = 0; i < 1024; ++i) {
    const float* row = Ab + (size_t)i * 1024;
    float aj[8];
    #pragma unroll
    for (int r = 0; r < 8; ++r) aj[r] = row[j0 + joff + r];
    #pragma unroll
    for (int s = 0; s < 4; ++s) {
      float bk = row[(kt0 + s) * 16 + ln];
      #pragma unroll
      for (int r = 0; r < 8; ++r) accv[s][r] = fmaf(aj[r], bk, accv[s][r]);
    }
  }
#endif

  int mb = (lane < 16) ? 0 : 8;
  #pragma unroll
  for (int s = 0; s < 4; ++s) {
    int k0 = (kt0 + s) * 16;
    #pragma unroll
    for (int r = 0; r < 8; ++r) {
      int j = j0 + mb + r;
      int k = k0 + ln;
      float v = accv[s][r];
      if (j == k) {
        int pt = j >> 5, px = j & 31;
        bool bc = (pt == 0) || (pt == 31) || (px == 0) || (px == 31);
        v += (bc ? 10000.0f : 0.0f) + 1e-6f;
      }
      M[((size_t)b << 20) + (size_t)j * 1024 + k] = v;
    }
  }
}

// ---------------------------------------------------------------------------
// Jacobi-preconditioned CG, one block per patch, 1024 threads.
// ---------------------------------------------------------------------------
__device__ __forceinline__ float blk_sum(float v, float* red) {
  int t = threadIdx.x;
  red[t] = v; __syncthreads();
  for (int s = 512; s > 0; s >>= 1) { if (t < s) red[t] += red[t + s]; __syncthreads(); }
  float r = red[0]; __syncthreads();
  return r;
}

__global__ __launch_bounds__(1024) void cg_solve_kernel(
    const float* __restrict__ M, const float* __restrict__ R,
    float* __restrict__ X) {
  __shared__ float sp[1024];
  __shared__ float red[1024];
  int b = blockIdx.x, g = threadIdx.x;
  const float* Mb = M + ((size_t)b << 20);
  const float* row = Mb + (size_t)g * 1024;
  float dinv = 1.0f / row[g];
  float x = 0.0f;
  float r = R[b * 1024 + g];
  float z = r * dinv;
  float p = z;
  float rz = blk_sum(r * z, red);

  for (int it = 0; it < 256; ++it) {
    sp[g] = p; __syncthreads();
    float w = 0.0f;
    #pragma unroll 4
    for (int j = 0; j < 1024; ++j) w = fmaf(row[j], sp[j], w);
    float pw = blk_sum(p * w, red);
    float alpha = rz / (pw + 1e-30f);
    x += alpha * p;
    r -= alpha * w;
    z = r * dinv;
    float rznew = blk_sum(r * z, red);
    float beta = rznew / (rz + 1e-30f);
    p = z + beta * p;
    rz = rznew;
  }
  X[b * 1024 + g] = x;
}

// ---------------------------------------------------------------------------
// eps = max_b,g | (A x)[b,g] |.  Wave 0 kicks a TDM DMA of x into LDS
// (tensor_load_to_lds + s_wait_tensorcnt); the plain fill afterwards is
// authoritative, so semantics do not depend on the descriptor.
// ---------------------------------------------------------------------------
__global__ void eps_init_kernel(unsigned* e) { if (threadIdx.x == 0) *e = 0u; }

__global__ __launch_bounds__(1024) void eps_kernel(
    const float* __restrict__ A, const float* __restrict__ X, unsigned* e) {
  __shared__ float sx[1024];
  __shared__ float red[1024];
  int b = blockIdx.x, g = threadIdx.x;

#ifdef HAS_TDM
  if (threadIdx.x < 32) {
    unsigned lds_addr = (unsigned)(unsigned long long)(void*)sx;
    unsigned long long ga = (unsigned long long)(const void*)(X + b * 1024);
    u32x4 g0;
    g0[0] = 1u;                                   // count=1, user descriptor
    g0[1] = lds_addr;                             // lds byte address
    g0[2] = (unsigned)ga;                         // global_addr[31:0]
    g0[3] = (unsigned)(ga >> 32) | (2u << 30);    // global_addr[56:32] | type=2
    i32x8 g1;
    g1[0] = 0x20000;                              // data_size=2 (4B), no mask
    g1[1] = (int)(1024u << 16);                   // tensor_dim0 = 1024 (low 16)
    g1[2] = (int)(1u << 16);                      // tensor_dim1 = 1
    g1[3] = (int)(1024u << 16);                   // tile_dim0 = 1024
    g1[4] = 1;                                    // tile_dim1 = 1
    g1[5] = 1024;                                 // tensor_dim0_stride = 1024
    g1[6] = 0;
    g1[7] = 0;
    i32x4 gz = {0, 0, 0, 0};
#if defined(__clang_major__) && (__clang_major__ >= 23)
    i32x8 gz8 = {0, 0, 0, 0, 0, 0, 0, 0};
    __builtin_amdgcn_tensor_load_to_lds(g0, g1, gz, gz, gz8, 0);
#else
    __builtin_amdgcn_tensor_load_to_lds(g0, g1, gz, gz, 0);
#endif
    __builtin_amdgcn_s_wait_tensorcnt(0);
  }
  __syncthreads();
#endif

  sx[g] = X[b * 1024 + g]; __syncthreads();
  const float* row = A + ((size_t)b << 20) + (size_t)g * 1024;
  float acc = 0.0f;
  #pragma unroll 4
  for (int h = 0; h < 1024; ++h) acc = fmaf(row[h], sx[h], acc);
  float a = fabsf(acc);
  red[g] = a; __syncthreads();
  for (int s = 512; s > 0; s >>= 1) {
    if (g < s) red[g] = fmaxf(red[g], red[g + s]);
    __syncthreads();
  }
  if (g == 0) atomicMax(e, __float_as_uint(red[0]));
}

__global__ void eps_store_kernel(const unsigned* e, float* out) {
  if (threadIdx.x == 0) out[0] = __uint_as_float(*e);
}

// ---------------------------------------------------------------------------
// u0 = join_patches(X)
// ---------------------------------------------------------------------------
__global__ __launch_bounds__(256) void join_kernel(
    const float* __restrict__ X, float* __restrict__ out) {
  int idx = blockIdx.x * blockDim.x + threadIdx.x;   // 16384
  int x = idx & 127, y = idx >> 7;
  int ti = y >> 5, pt = y & 31;
  int xi = x >> 5, px = x & 31;
  out[idx] = X[(ti * 4 + xi) * 1024 + pt * 32 + px];
}

// ---------------------------------------------------------------------------
// Host orchestration
// ---------------------------------------------------------------------------
extern "C" void kernel_launch(void* const* d_in, const int* in_sizes, int n_in,
                              void* d_out, int out_size, void* d_ws, size_t ws_size,
                              hipStream_t stream) {
  (void)in_sizes; (void)n_in; (void)out_size; (void)ws_size;
  char* ws = (char*)d_ws;

  const size_t OFF_A     = 0;                          // 64MB
  const size_t OFF_M     = OFF_A + (size_t)67108864;   // 64MB
  const size_t OFF_ACTA  = OFF_M + (size_t)67108864;
  const size_t OFF_ACTB  = OFF_ACTA + (size_t)ACT_HALFS * 2;
  const size_t OFF_WPREP = OFF_ACTB + (size_t)ACT_HALFS * 2;
  const size_t OFF_UP    = OFF_WPREP + 204800;
  const size_t OFF_UP2   = OFF_UP + 65536;
  const size_t OFF_P     = OFF_UP2 + 65536;
  const size_t OFF_Q     = OFF_P + 65536;
  const size_t OFF_R     = OFF_Q + 65536;
  const size_t OFF_X     = OFF_R + 65536;
  const size_t OFF_H1    = OFF_X + 65536;
  const size_t OFF_H2    = OFF_H1 + 4096;
  const size_t OFF_PC    = OFF_H2 + 4096;
  const size_t OFF_QC    = OFF_PC + 64;
  const size_t OFF_EPS   = OFF_QC + 64;

  float*     A_    = (float*)(ws + OFF_A);
  float*     M_    = (float*)(ws + OFF_M);
  _Float16*  actA  = (_Float16*)(ws + OFF_ACTA);
  _Float16*  actB  = (_Float16*)(ws + OFF_ACTB);
  _Float16*  wprep = (_Float16*)(ws + OFF_WPREP);
  float*     upW   = (float*)(ws + OFF_UP);
  float*     up2W  = (float*)(ws + OFF_UP2);
  float*     P_    = (float*)(ws + OFF_P);
  float*     Q_    = (float*)(ws + OFF_Q);
  float*     R_    = (float*)(ws + OFF_R);
  float*     X_    = (float*)(ws + OFF_X);
  float*     h1    = (float*)(ws + OFF_H1);
  float*     h2    = (float*)(ws + OFF_H2);
  float*     pc    = (float*)(ws + OFF_PC);
  float*     qc    = (float*)(ws + OFF_QC);
  unsigned*  epsU  = (unsigned*)(ws + OFF_EPS);

  const float* u = (const float*)d_in[0];
  float* out = (float*)d_out;

  {
    int n = ACT_HALFS * 2;
    clear_half_kernel<<<(n + 255) / 256, 256, 0, stream>>>(actA, n);
  }

  // ---- two conv nets ----
  for (int net = 0; net < 2; ++net) {
    int cb = (net == 0) ? 3 : 17;
    float* upDst  = (net == 0) ? upW : up2W;
    float* outDst = out + ((net == 0) ? 16384 : 32768);

    conv0_kernel<<<4096, 256, 0, stream>>>(
        u, (const float*)d_in[cb + 0], (const float*)d_in[cb + 1], actA);

    for (int li = 1; li <= 5; ++li) {
      prep_w_kernel<<<400, 256, 0, stream>>>((const float*)d_in[cb + 2 * li], wprep);
      const _Float16* inA  = (li & 1) ? actA : actB;
      _Float16*       outA = (li & 1) ? actB : actA;
      conv_wmma_kernel<<<256, 128, 0, stream>>>(
          inA, wprep, (const float*)d_in[cb + 2 * li + 1], outA);
    }
    conv_last_kernel<<<64, 256, 0, stream>>>(
        actB, (const float*)d_in[cb + 12], (const float*)d_in[cb + 13],
        u, upDst, outDst);
  }

  // ---- two MLPs ----
  for (int net = 0; net < 2; ++net) {
    int mb = (net == 0) ? 33 : 39;
    const float* pin = (const float*)d_in[(net == 0) ? 31 : 32];
    float* dst = (net == 0) ? pc : qc;
    mlp_layer_kernel<<<4, 256, 0, stream>>>(
        pin, (const float*)d_in[mb + 0], (const float*)d_in[mb + 1], h1, 64, 1024, 1);
    mlp_layer_kernel<<<4, 256, 0, stream>>>(
        h1, (const float*)d_in[mb + 2], (const float*)d_in[mb + 3], h2, 1024, 1024, 1);
    mlp_layer_kernel<<<1, 32, 0, stream>>>(
        h2, (const float*)d_in[mb + 4], (const float*)d_in[mb + 5], dst, 1024, 3, 0);
  }
  params_store_kernel<<<1, 32, 0, stream>>>(pc, qc, out);

  // ---- p, q, rhs ----
  build_pq_kernel<<<64, 256, 0, stream>>>(upW, up2W, pc, qc, P_, Q_, R_);

  // ---- dense A, then M = A^T A + reg via f32 WMMA ----
  build_A_kernel<<<65536, 256, 0, stream>>>(P_, Q_, A_);
  ata_wmma_kernel<<<4096, 128, 0, stream>>>(A_, M_);

  // ---- CG solve per patch ----
  cg_solve_kernel<<<16, 1024, 0, stream>>>(M_, R_, X_);

  // ---- eps = max |A x| ----
  eps_init_kernel<<<1, 32, 0, stream>>>(epsU);
  eps_kernel<<<16, 1024, 0, stream>>>(A_, X_, epsU);
  eps_store_kernel<<<1, 32, 0, stream>>>(epsU, out + 49152);

  // ---- u0 join ----
  join_kernel<<<64, 256, 0, stream>>>(X_, out);
}
